// UniVariateSeq2Seq_15650860827305
// MI455X (gfx1250) — matrix operand order repaired and checked by reading, not compiled, not run on hardware
//
#include <hip/hip_runtime.h>

// ---------------- problem constants ----------------
#define HD    1024      // hidden size
#define BATCH 64
#define TENC  512
#define TDEC  128
#define NWG   64        // persistent workgroups
#define JC    16        // hidden columns owned per WG  (NWG*JC == HD)
#define BT    256       // threads per block (8 wave32)

// ---------------- WMMA vector types ----------------
typedef __attribute__((ext_vector_type(16))) __bf16 v16bf;
typedef __attribute__((ext_vector_type(8)))  __bf16 v8bf;
typedef __attribute__((ext_vector_type(8)))  float  v8f;

// ---------------- helpers ----------------
__device__ __forceinline__ unsigned short f2bf(float f) {
    unsigned u = __float_as_uint(f);
    u += 0x7FFFu + ((u >> 16) & 1u);          // round-to-nearest-even
    return (unsigned short)(u >> 16);
}
__device__ __forceinline__ float sigm(float x) {
    return 1.0f / (1.0f + __expf(-x));
}

// device-wide generation barrier (persistent kernel, NWG blocks)
__device__ __forceinline__ void grid_barrier(unsigned* bar) {
    __threadfence();          // publish this thread's global stores
    __syncthreads();
    if (threadIdx.x == 0) {
        unsigned gen = __hip_atomic_load(&bar[1], __ATOMIC_RELAXED,
                                         __HIP_MEMORY_SCOPE_AGENT);
        unsigned arrived = __hip_atomic_fetch_add(&bar[0], 1u, __ATOMIC_ACQ_REL,
                                                  __HIP_MEMORY_SCOPE_AGENT);
        if (arrived == (unsigned)(NWG - 1)) {
            __hip_atomic_store(&bar[0], 0u, __ATOMIC_RELAXED,
                               __HIP_MEMORY_SCOPE_AGENT);
            __hip_atomic_fetch_add(&bar[1], 1u, __ATOMIC_RELEASE,
                                   __HIP_MEMORY_SCOPE_AGENT);
        } else {
            while (__hip_atomic_load(&bar[1], __ATOMIC_ACQUIRE,
                                     __HIP_MEMORY_SCOPE_AGENT) == gen) {
                __builtin_amdgcn_s_sleep(2);
            }
        }
        __threadfence();      // acquire: invalidate caches before next reads
    }
    __syncthreads();
}

// ---------------- prep kernels ----------------
__global__ void cvt_f32_bf16(const float* __restrict__ src,
                             unsigned short* __restrict__ dst, int n) {
    int i = blockIdx.x * blockDim.x + threadIdx.x;
    if (i < n) dst[i] = f2bf(src[i]);
}

__global__ void init_state(unsigned short* __restrict__ h0,
                           unsigned* __restrict__ bar) {
    int i = blockIdx.x * blockDim.x + threadIdx.x;
    if (i < BATCH * HD) h0[i] = 0;        // encoder h0 = 0 (bf16 zero)
    if (i < 2) bar[i] = 0;                // barrier counter + generation
}

// ---------------- final output reduction ----------------
__global__ void reduce_out(const float* __restrict__ partials,  // [TDEC][NWG][BATCH]
                           const float* __restrict__ out_b,
                           float* __restrict__ out) {           // [BATCH][TDEC]
    int t = blockIdx.x;          // decoder step
    int b = threadIdx.x;         // batch row
    float s = out_b[0];
    #pragma unroll 4
    for (int w = 0; w < NWG; ++w)
        s += partials[((size_t)t * NWG + w) * BATCH + b];
    out[(size_t)b * TDEC + t] = s;
}

// ---------------- persistent seq2seq kernel ----------------
// shared memory layout (dynamic):
//   sW    : 4*JC*HD bf16 weight slice        131072 B
//   sGate : [4][BATCH][JC] f32                16384 B
//   sC    : [BATCH][JC]    f32                 4096 B
//   sPart : [BATCH][JC]    f32                 4096 B
//   sWih  : [4][JC] f32, sBias : [4][JC] f32, sOutW : [JC] f32
#define SM_W     0
#define SM_GATE  (131072)
#define SM_C     (SM_GATE + 16384)
#define SM_PART  (SM_C + 4096)
#define SM_WIH   (SM_PART + 4096)
#define SM_BIAS  (SM_WIH + 256)
#define SM_OUTW  (SM_BIAS + 256)
#define SM_BYTES (SM_OUTW + 64)

__global__ void __launch_bounds__(BT, 2)
seq2seq_persistent(const float* __restrict__ xin,          // [B][TENC]
                   const float* __restrict__ tgt,          // [B][TDEC]
                   const unsigned short* __restrict__ Wenc,// bf16 [4H][H]
                   const float* __restrict__ WihEnc,       // [4H]
                   const float* __restrict__ bEnc,         // [4H]
                   const unsigned short* __restrict__ Wdec,
                   const float* __restrict__ WihDec,
                   const float* __restrict__ bDec,
                   const float* __restrict__ outW,         // [H]
                   unsigned short* __restrict__ h0buf,     // bf16 [B][H] ping
                   unsigned short* __restrict__ h1buf,     // bf16 [B][H] pong
                   float* __restrict__ partials,           // [TDEC][NWG][B]
                   unsigned* __restrict__ bar) {
    extern __shared__ char smem[];
    unsigned short* sW    = (unsigned short*)(smem + SM_W);
    float*          sGate = (float*)(smem + SM_GATE);
    float*          sC    = (float*)(smem + SM_C);
    float*          sPart = (float*)(smem + SM_PART);
    float*          sWih  = (float*)(smem + SM_WIH);
    float*          sBias = (float*)(smem + SM_BIAS);
    float*          sOutW = (float*)(smem + SM_OUTW);

    const int wg    = blockIdx.x;          // owns hidden cols [wg*JC, wg*JC+JC)
    const int wave  = threadIdx.x >> 5;
    const int lane  = threadIdx.x & 31;
    // wave -> unique (M-tile, gate-pair): A loaded once per M-tile per gate-pair,
    // duplicated operand is the LDS-resident B fragment (cheap to re-read).
    const int mt    = wave >> 1;           // M-tile 0..3 (batch rows mt*16..)
    const int g0    = (wave & 1) * 2;      // gates {g0, g0+1}
    const int jl    = lane & 15;           // N index within tile
    const int kA    = (lane >> 4) << 4;    // A-frag K sub-offset (0 / 16)
    const int kB    = (lane >> 4) << 4;    // B-frag K sub-offset (0 / 16)

    // zero cell state
    for (int u = threadIdx.x; u < BATCH * JC; u += BT) sC[u] = 0.0f;

    for (int phase = 0; phase < 2; ++phase) {
        const bool isDec = (phase == 1);
        const unsigned short* Wg = isDec ? Wdec : Wenc;
        const float* wih = isDec ? WihDec : WihEnc;
        const float* bb  = isDec ? bDec   : bEnc;
        const int    nT  = isDec ? TDEC   : TENC;

        // ---- stage this WG's weight slice into LDS (once per phase) ----
        for (int u = threadIdx.x; u < 64 * 128; u += BT) {   // 16B units
            int rowL = u >> 7;                 // 0..63  (gate*16 + jlocal)
            int kk   = (u & 127) << 3;         // 8 bf16 per 16B
            int g    = rowL >> 4, jloc = rowL & 15;
            size_t R = (size_t)g * HD + (size_t)wg * JC + jloc;
            *(int4*)(sW + rowL * HD + kk) = *(const int4*)(Wg + R * HD + kk);
        }
        if (threadIdx.x < 64) {
            int g = threadIdx.x >> 4, jloc = threadIdx.x & 15;
            int R = g * HD + wg * JC + jloc;
            sWih[threadIdx.x]  = wih[R];
            sBias[threadIdx.x] = bb[R];
        }
        if (isDec && threadIdx.x < JC) sOutW[threadIdx.x] = outW[wg * JC + threadIdx.x];
        __syncthreads();

        for (int t = 0; t < nT; ++t) {
            const int step = phase * TENC + t;          // global step parity
            const unsigned short* hsrc = (step & 1) ? h1buf : h0buf;
            unsigned short*       hdst = (step & 1) ? h0buf : h1buf;

            // ---- gates = h @ Whh.T  (bf16 WMMA, f32 accum) ----
            const __bf16* hb = (const __bf16*)hsrc;
            const __bf16* wb = (const __bf16*)sW;
            const int rowA  = (mt * 16 + (lane & 15)) * HD;
            const int rowB0 = (g0 * 16 + jl) * HD;
            const int rowB1 = rowB0 + 16 * HD;

            v8f acc0 = {}; v8f acc1 = {};
            #pragma unroll 4
            for (int k0 = 0; k0 < HD; k0 += 32) {
                // B fragments from LDS (K split across half-waves, contiguous 32B)
                v16bf b0 = *(const v16bf*)(wb + rowB0 + k0 + kB);
                v16bf b1 = *(const v16bf*)(wb + rowB1 + k0 + kB);
                // A fragment from global h (lane m: K 0-15; m+16: 16-31)
                v8bf alo = *(const v8bf*)(hb + rowA + k0 + kA);
                v8bf ahi = *(const v8bf*)(hb + rowA + k0 + kA + 8);
                v16bf a;
                #pragma unroll
                for (int i = 0; i < 8; ++i) { a[i] = alo[i]; a[i + 8] = ahi[i]; }
                acc0 = __builtin_amdgcn_wmma_f32_16x16x32_bf16(
                           false, a, false, b0, (short)0, acc0, false, false);
                acc1 = __builtin_amdgcn_wmma_f32_16x16x32_bf16(
                           false, a, false, b1, (short)0, acc1, false, false);
            }

            // ---- spill accumulators to LDS gate buffer ----
            // D layout: VGPR v, lanes 0-15 -> M=v, lanes 16-31 -> M=v+8; N=lane%16
            {
                const int rb = (lane < 16) ? 0 : 8;
                float* gd0 = sGate + ((g0 * BATCH) + mt * 16 + rb) * JC + jl;
                float* gd1 = gd0 + BATCH * JC;          // gate g0+1
                #pragma unroll
                for (int v = 0; v < 8; ++v) gd0[v * JC] = acc0[v];
                #pragma unroll
                for (int v = 0; v < 8; ++v) gd1[v * JC] = acc1[v];
            }
            __syncthreads();

            // ---- elementwise LSTM cell update for owned columns ----
            #pragma unroll
            for (int q = 0; q < 4; ++q) {
                int idx = threadIdx.x + q * BT;       // 0..1023 = b*16 + j
                int b = idx >> 4, j = idx & 15;
                float xv = isDec
                    ? (t == 0 ? xin[b * TENC + (TENC - 1)] : tgt[b * TDEC + (t - 1)])
                    : xin[b * TENC + t];
                float iv = sGate[(0 * BATCH + b) * JC + j] + xv * sWih[ 0 + j] + sBias[ 0 + j];
                float fv = sGate[(1 * BATCH + b) * JC + j] + xv * sWih[16 + j] + sBias[16 + j];
                float gv = sGate[(2 * BATCH + b) * JC + j] + xv * sWih[32 + j] + sBias[32 + j];
                float ov = sGate[(3 * BATCH + b) * JC + j] + xv * sWih[48 + j] + sBias[48 + j];
                float c  = sigm(fv) * sC[idx] + sigm(iv) * tanhf(gv);
                sC[idx]  = c;
                float hn = sigm(ov) * tanhf(c);
                hdst[b * HD + wg * JC + j] = f2bf(hn);
                if (isDec) sPart[idx] = hn * sOutW[j];
            }

            // ---- decoder projection partials (deterministic) ----
            if (isDec) {
                __syncthreads();
                if (threadIdx.x < BATCH) {
                    float s = 0.0f;
                    #pragma unroll
                    for (int j = 0; j < JC; ++j) s += sPart[threadIdx.x * JC + j];
                    partials[((size_t)t * NWG + wg) * BATCH + threadIdx.x] = s;
                }
            }

            grid_barrier(bar);   // publish h_{t}, wait for all WGs
        }
        __syncthreads();         // before overwriting sW with decoder weights
    }
}

// ---------------- workspace layout ----------------
#define OFF_BAR   ((size_t)0)
#define OFF_H0    ((size_t)256)
#define OFF_H1    (OFF_H0 + (size_t)BATCH * HD * 2)
#define OFF_WENC  (OFF_H1 + (size_t)BATCH * HD * 2)
#define OFF_WDEC  (OFF_WENC + (size_t)4 * HD * HD * 2)
#define OFF_PART  (OFF_WDEC + (size_t)4 * HD * HD * 2)

extern "C" void kernel_launch(void* const* d_in, const int* in_sizes, int n_in,
                              void* d_out, int out_size, void* d_ws, size_t ws_size,
                              hipStream_t stream) {
    (void)in_sizes; (void)n_in; (void)out_size; (void)ws_size;

    const float* xin     = (const float*)d_in[0];   // [B,TENC,1]
    const float* tgt     = (const float*)d_in[1];   // [B,TDEC,1]
    const float* enc_Wih = (const float*)d_in[2];   // [4H,1]
    const float* enc_Whh = (const float*)d_in[3];   // [4H,H]
    const float* enc_b   = (const float*)d_in[4];   // [4H]
    const float* dec_Wih = (const float*)d_in[5];
    const float* dec_Whh = (const float*)d_in[6];
    const float* dec_b   = (const float*)d_in[7];
    const float* out_W   = (const float*)d_in[8];   // [1,H]
    const float* out_b   = (const float*)d_in[9];   // [1]
    float* out = (float*)d_out;                     // [B,TDEC,1]

    char* ws = (char*)d_ws;
    unsigned*       bar   = (unsigned*)(ws + OFF_BAR);
    unsigned short* h0    = (unsigned short*)(ws + OFF_H0);
    unsigned short* h1    = (unsigned short*)(ws + OFF_H1);
    unsigned short* wenc  = (unsigned short*)(ws + OFF_WENC);
    unsigned short* wdec  = (unsigned short*)(ws + OFF_WDEC);
    float*          parts = (float*)(ws + OFF_PART);

    // 1) init h0 (= zeros) and barrier state
    init_state<<<(BATCH * HD + 255) / 256, 256, 0, stream>>>(h0, bar);

    // 2) convert recurrent weights to bf16 (L2-resident thereafter)
    const int NW = 4 * HD * HD;
    cvt_f32_bf16<<<(NW + 255) / 256, 256, 0, stream>>>(enc_Whh, wenc, NW);
    cvt_f32_bf16<<<(NW + 255) / 256, 256, 0, stream>>>(dec_Whh, wdec, NW);

    // 3) persistent recurrent kernel: 64 WGs, one grid barrier per timestep
    seq2seq_persistent<<<NWG, BT, SM_BYTES, stream>>>(
        xin, tgt, wenc, enc_Wih, enc_b, wdec, dec_Wih, dec_b, out_W,
        h0, h1, parts, bar);

    // 4) deterministic reduction of decoder projection partials
    reduce_out<<<TDEC, BATCH, 0, stream>>>(parts, out_b, out);
}
